// DeepseekV3Gate_60894046323248
// MI455X (gfx1250) — compile-verified, matrix operand order checked by reading, not run on hardware
//
#include <hip/hip_runtime.h>
#include <hip/hip_bf16.h>
#include <math.h>

// ---------------------------------------------------------------------------
// DeepseekV3 gate: logits = x @ W_gate (bf16 WMMA, f32 accum), then sigmoid /
// grouped top-k routing.  T=16384, H=7168, E=256, 8 groups of 32, keep top-4
// groups, top-8 experts, sigmoid-normalized weights * 2.5.
//
// Roofline: x (470 MB fp32) is the only mandatory HBM stream (~20 us @ 23.3
// TB/s). Each wave owns a 16-token x 256-expert slab (x read exactly once).
// The shared B k-tile (16 KB) is staged through double-buffered LDS once per
// block; B fragments are read from LDS in groups of 4 (one dscnt wait per 4
// WMMAs) and the A fragment for kt+1 is loaded a full compute phase ahead.
// ---------------------------------------------------------------------------

#define T_TOK   16384
#define HID     7168
#define NEXP    256
#define NGRP    8
#define TOPG    4
#define TOPK    8
#define KSTEP   32            // bf16 WMMA K
#define KTILES  (HID / KSTEP) // 224
#define NTILES  (NEXP / 16)   // 16
#define TILE_DW (NTILES * 32 * 8)                 // 4096 dwords = 16 KB / k-tile
#define WB_DWORDS (KTILES * TILE_DW)              // packed W: dwords
#define WB_BYTES  (WB_DWORDS * 4)                 // 3,670,016 bytes

typedef __attribute__((ext_vector_type(16))) __bf16 v16bf;
typedef __attribute__((ext_vector_type(8)))  float  v8f;
typedef __attribute__((ext_vector_type(4)))  float  f32x4;

__device__ __forceinline__ unsigned f2bf(float f) {
    unsigned u = __float_as_uint(f);
    unsigned r = u + 0x7fffu + ((u >> 16) & 1u);   // round-to-nearest-even
    return r >> 16;
}

// ---------------------------------------------------------------------------
// Kernel 0: repack W_gate [HID x NEXP] f32 row-major into bf16 WMMA B-frags.
// Frag (kt, ct): lane holds col = ct*16 + (lane&15); K chunk base
// kb = kt*32 + (lane<16 ? 0 : 8); 16 bf16 values = K[kb..kb+7], K[kb+16..kb+23]
// stored as 8 dwords contiguous per lane.
// ---------------------------------------------------------------------------
__global__ __launch_bounds__(256) void pack_w(const float* __restrict__ W,
                                              unsigned* __restrict__ wB) {
    int id = blockIdx.x * blockDim.x + threadIdx.x;   // KTILES*NTILES*32 threads
    if (id >= KTILES * NTILES * 32) return;
    int lane = id & 31;
    int ct   = (id >> 5) & (NTILES - 1);
    int kt   = id >> 9;
    int col  = ct * 16 + (lane & 15);
    int kb   = kt * KSTEP + ((lane & 16) ? 8 : 0);
    unsigned out[8];
#pragma unroll
    for (int p = 0; p < 8; ++p) {
        int kk = kb + (p < 4 ? 2 * p : 16 + 2 * (p - 4));
        unsigned lo = f2bf(W[kk * NEXP + col]);
        unsigned hi = f2bf(W[(kk + 1) * NEXP + col]);
        out[p] = lo | (hi << 16);
    }
    unsigned* dst = wB + (size_t)((kt * NTILES + ct) * 32 + lane) * 8;
    *(uint4*)(dst)     = make_uint4(out[0], out[1], out[2], out[3]);
    *(uint4*)(dst + 4) = make_uint4(out[4], out[5], out[6], out[7]);
}

// ---------------------------------------------------------------------------
// Kernel 1: block = 8 waves = 128 tokens x 256 experts. Wave w owns row-tile w
// and ALL 16 column tiles. B k-tile staged via double-buffered LDS per block;
// A fragment prefetched one k-iteration ahead into fp32 registers.
// ---------------------------------------------------------------------------
__global__ __launch_bounds__(256) void gemm_bf16(const float* __restrict__ x,
                                                 const unsigned* __restrict__ wB,
                                                 float* __restrict__ logits) {
    __shared__ __align__(16) unsigned lbuf[2][TILE_DW];   // 2 x 16 KB

    const int tid  = threadIdx.x;
    const int wave = tid >> 5;
    const int lane = tid & 31;
    const int m16  = lane & 15;               // M (A) / N (B,C) within tile
    const int hi   = (lane & 16) ? 8 : 0;     // lane-half K offset
    const int rowA = blockIdx.x * 128 + wave * 16 + m16;
    const float* __restrict__ xrow = x + (size_t)rowA * HID;

    v8f acc[NTILES];
#pragma unroll
    for (int i = 0; i < NTILES; ++i)
        acc[i] = (v8f){0.f,0.f,0.f,0.f,0.f,0.f,0.f,0.f};

    // ---- prologue: stage B k-tile 0 into lbuf[0]; A regs for kt = 0 --------
    {
        const uint4* src = (const uint4*)wB + tid * 4;
        uint4 r0 = src[0], r1 = src[1], r2 = src[2], r3 = src[3];
        uint4* dst = (uint4*)&lbuf[0][tid * 16];
        dst[0] = r0; dst[1] = r1; dst[2] = r2; dst[3] = r3;
    }
    f32x4 a0 = *(const f32x4*)(xrow + hi);
    f32x4 a1 = *(const f32x4*)(xrow + hi + 4);
    f32x4 a2 = *(const f32x4*)(xrow + hi + 16);
    f32x4 a3 = *(const f32x4*)(xrow + hi + 20);
    __syncthreads();

    for (int kt = 0; kt < KTILES; ++kt) {
        const int cur  = kt & 1;
        const bool more = (kt + 1 < KTILES);

        // ---- issue global loads for next B k-tile (overlaps with compute) --
        uint4 r0, r1, r2, r3;
        if (more) {
            const uint4* src = (const uint4*)(wB + (size_t)(kt + 1) * TILE_DW) + tid * 4;
            r0 = src[0]; r1 = src[1]; r2 = src[2]; r3 = src[3];
        }

        // ---- convert current A (loaded last iteration) to bf16 frag --------
        float av[16] = { a0.x, a0.y, a0.z, a0.w, a1.x, a1.y, a1.z, a1.w,
                         a2.x, a2.y, a2.z, a2.w, a3.x, a3.y, a3.z, a3.w };
        union { unsigned u[8]; v16bf v; } A;
#pragma unroll
        for (int p = 0; p < 8; ++p)
            A.u[p] = f2bf(av[2 * p]) | (f2bf(av[2 * p + 1]) << 16);

        // ---- prefetch A for kt+1 a full compute phase ahead ----------------
        if (more) {
            const int kn = (kt + 1) * KSTEP;
            __builtin_prefetch(xrow + kn + KSTEP, 0, 0);
            a0 = *(const f32x4*)(xrow + kn + hi);
            a1 = *(const f32x4*)(xrow + kn + hi + 4);
            a2 = *(const f32x4*)(xrow + kn + hi + 16);
            a3 = *(const f32x4*)(xrow + kn + hi + 20);
        }

        // ---- 16 B fragments from LDS, grouped 4-wide: 1 wait per 4 WMMAs ---
        const unsigned* lb = &lbuf[cur][lane * 8];
#pragma unroll
        for (int nt4 = 0; nt4 < NTILES; nt4 += 4) {
            union { uint4 q[8]; v16bf v[4]; } B;
#pragma unroll
            for (int f = 0; f < 4; ++f) {
                B.q[2 * f]     = *(const uint4*)(lb + (nt4 + f) * 256);
                B.q[2 * f + 1] = *(const uint4*)(lb + (nt4 + f) * 256 + 4);
            }
#pragma unroll
            for (int f = 0; f < 4; ++f)
                acc[nt4 + f] = __builtin_amdgcn_wmma_f32_16x16x32_bf16(
                    false, A.v, false, B.v[f], (short)0, acc[nt4 + f], false, false);
        }

        // ---- publish next B k-tile ----------------------------------------
        if (more) {
            uint4* dst = (uint4*)&lbuf[cur ^ 1][tid * 16];
            dst[0] = r0; dst[1] = r1; dst[2] = r2; dst[3] = r3;
            __syncthreads();
        }
    }

    // ---- store C: lane holds N=lane&15, VGPR v -> M = v + (lane>=16 ? 8 : 0)
    const int mbase = blockIdx.x * 128 + wave * 16 + hi;
#pragma unroll
    for (int nt = 0; nt < NTILES; ++nt) {
#pragma unroll
        for (int v = 0; v < 8; ++v)
            logits[(size_t)(mbase + v) * NEXP + nt * 16 + m16] = acc[nt][v];
    }
}

// ---------------------------------------------------------------------------
// Kernel 2: routing, one wave32 per token. Lane l owns experts 8l..8l+7.
// ---------------------------------------------------------------------------
__global__ __launch_bounds__(256) void route(const float* __restrict__ logits,
                                             const float* __restrict__ bias,
                                             float* __restrict__ out_w,
                                             float* __restrict__ out_i) {
    const int tok  = blockIdx.x * 8 + (threadIdx.x >> 5);
    const int lane = threadIdx.x & 31;
    const float NINF = -__builtin_inff();

    float sig[8], sc[8];
#pragma unroll
    for (int j = 0; j < 8; ++j) {
        float lg = logits[(size_t)tok * NEXP + lane * 8 + j];
        float sg = 1.0f / (1.0f + __expf(-lg));
        sig[j] = sg;
        sc[j]  = sg + bias[lane * 8 + j];
    }

    // per-lane top-2 of biased scores
    float a1 = NINF, a2 = NINF;
#pragma unroll
    for (int j = 0; j < 8; ++j) {
        float s = sc[j];
        if (s > a1) { a2 = a1; a1 = s; } else if (s > a2) { a2 = s; }
    }
    // merge over the 4 lanes of each group (group = lane>>2)
#pragma unroll
    for (int off = 1; off <= 2; off <<= 1) {
        float b1 = __shfl_xor(a1, off, 32);
        float b2 = __shfl_xor(a2, off, 32);
        float m1 = fmaxf(a1, b1);
        float m2 = fmaxf(fminf(a1, b1), fmaxf(a2, b2));
        a1 = m1; a2 = m2;
    }
    float gscore = a1 + a2;

    // gather 8 group scores, pick top-4 groups (lowest index on ties)
    float gs[8];
#pragma unroll
    for (int g = 0; g < 8; ++g) gs[g] = __shfl(gscore, g * 4, 32);
    unsigned gmask = 0;
#pragma unroll
    for (int k = 0; k < TOPG; ++k) {
        float best = NINF; int bg = 0;
#pragma unroll
        for (int g = 0; g < 8; ++g)
            if (!((gmask >> g) & 1u) && gs[g] > best) { best = gs[g]; bg = g; }
        gmask |= 1u << bg;
    }
    if (!((gmask >> (lane >> 2)) & 1u)) {
#pragma unroll
        for (int j = 0; j < 8; ++j) sc[j] = NINF;
    }

    // top-8 experts: 8 wave-wide argmax reductions, lowest-index tie-break
    unsigned usedMask = 0;
    float wsum = 0.0f, myw = 0.0f;
    int myi = 0;
#pragma unroll
    for (int k = 0; k < TOPK; ++k) {
        float bs = NINF, bsig = 0.0f; int bj = -1;
#pragma unroll
        for (int j = 0; j < 8; ++j)
            if (!((usedMask >> j) & 1u) && sc[j] > bs) { bs = sc[j]; bsig = sig[j]; bj = j; }
        int bidx = (bj < 0) ? 0x7fffffff : lane * 8 + bj;
#pragma unroll
        for (int off = 16; off >= 1; off >>= 1) {
            float os = __shfl_xor(bs, off, 32);
            float og = __shfl_xor(bsig, off, 32);
            int   oi = __shfl_xor(bidx, off, 32);
            if (os > bs || (os == bs && oi < bidx)) { bs = os; bsig = og; bidx = oi; }
        }
        wsum += bsig;
        if ((bidx >> 3) == lane) usedMask |= 1u << (bidx & 7);
        if (lane == k) { myw = bsig; myi = bidx; }
    }

    const float scale = 2.5f / wsum;
    if (lane < TOPK) {
        out_w[(size_t)tok * TOPK + lane] = myw * scale;
        out_i[(size_t)tok * TOPK + lane] = (float)myi;
    }
}

// ---------------------------------------------------------------------------
extern "C" void kernel_launch(void* const* d_in, const int* in_sizes, int n_in,
                              void* d_out, int out_size, void* d_ws, size_t ws_size,
                              hipStream_t stream) {
    const float* x    = (const float*)d_in[0];   // [T_TOK, HID]
    const float* W    = (const float*)d_in[1];   // [HID, NEXP]
    const float* bias = (const float*)d_in[2];   // [NEXP]

    unsigned* wB   = (unsigned*)d_ws;                          // 3.67 MB packed W
    float* logits  = (float*)((char*)d_ws + WB_BYTES);         // 16.8 MB logits
    float* out_w   = (float*)d_out;                            // [T_TOK, 8] weights
    float* out_i   = out_w + (size_t)T_TOK * TOPK;             // [T_TOK, 8] indices

    pack_w<<<(KTILES * NTILES * 32 + 255) / 256, 256, 0, stream>>>(W, wB);
    gemm_bf16<<<T_TOK / 128, 256, 0, stream>>>(x, wB, logits);
    route<<<T_TOK / 8, 256, 0, stream>>>(logits, bias, out_w, out_i);
}
